// SpectralMamba_89816356094272
// MI455X (gfx1250) — matrix operand order repaired and checked by reading, not compiled, not run on hardware
//
#include <hip/hip_runtime.h>
#include <hip/hip_bf16.h>

typedef __attribute__((ext_vector_type(16))) __bf16 v16bf;
typedef __attribute__((ext_vector_type(8)))  float  v8f;

#define BB 16
#define LL 4096
#define DM 64
#define DIN 128
#define NST 16
#define NTOK (BB*LL)          // 65536
#define NCHUNK 64
#define TC 64                 // chunk length

__device__ __forceinline__ float silu_f(float v) { return v / (1.f + __expf(-v)); }
__device__ __forceinline__ float softplus_f(float v) { return (v > 20.f) ? v : log1pf(__expf(v)); }

// ---------------------------------------------------------------------------
// Tensor Data Mover: 1-D contiguous copy global -> LDS (one wave issues).
// D# per CDNA5 ISA:
//  group0: count=1 | lds_addr | global_addr[56:0] | type=2
//  group1: data_size=4B, tensor_dim0=tile_dim0=nfloats, tensor_dim1=tile_dim1=1
// ---------------------------------------------------------------------------
#if __has_builtin(__builtin_amdgcn_tensor_load_to_lds)
#define HAVE_TDM 1
typedef __attribute__((ext_vector_type(4))) unsigned int u32x4_t;
typedef __attribute__((ext_vector_type(8))) int          i32x8_t;
typedef __attribute__((ext_vector_type(4))) int          i32x4_t;

__device__ __forceinline__ void tdm_load_1d(void* lds_dst, const void* gsrc,
                                            unsigned nfloats) {
    unsigned lds_lo = (unsigned)(uintptr_t)lds_dst;   // flat-LDS aperture: offset in [31:0]
    unsigned long long ga = (unsigned long long)(uintptr_t)gsrc;
    u32x4_t g0;
    g0[0] = 1u;                                             // count=1, user mode
    g0[1] = lds_lo;                                         // lds_addr
    g0[2] = (unsigned)(ga & 0xFFFFFFFFu);                   // global_addr[31:0]
    g0[3] = (unsigned)((ga >> 32) & 0x01FFFFFFu) | 0x80000000u; // [56:32] | type=2
    i32x8_t g1;
    g1[0] = 0x00020000;                                     // data_size=2 (4 bytes)
    g1[1] = (int)((nfloats & 0xFFFFu) << 16);               // tensor_dim0 lo (bits 79:48)
    g1[2] = (int)(((nfloats >> 16) & 0xFFFFu) | (1u << 16));// tensor_dim0 hi | tensor_dim1=1
    g1[3] = (int)((nfloats & 0xFFFFu) << 16);               // tile_dim0 (bits 127:112)
    g1[4] = 1;                                              // tile_dim1=1, tile_dim2=0
    g1[5] = (int)nfloats;                                   // tensor_dim0_stride lo
    g1[6] = 0;
    g1[7] = 0;
    i32x4_t zz = {0, 0, 0, 0};
#if defined(__clang_major__) && (__clang_major__ >= 23)
    i32x8_t g4 = {0, 0, 0, 0, 0, 0, 0, 0};
    __builtin_amdgcn_tensor_load_to_lds(g0, g1, zz, zz, g4, 0);
#else
    __builtin_amdgcn_tensor_load_to_lds(g0, g1, zz, zz, 0);
#endif
}
#endif

// ---------------------------------------------------------------------------
// K1: in_proj  xz[m, 0:256] = x[m, 0:64] @ W^T   (bf16 WMMA, fp32 acc)
// grid (4096, 4), block 128 (4 waves); wave -> one 16x16 output tile
// ---------------------------------------------------------------------------
__global__ void k_in_proj(const float* __restrict__ x, const float* __restrict__ w,
                          float* __restrict__ xi, float* __restrict__ z) {
    const int lane = threadIdx.x & 31;
    const int wv   = threadIdx.x >> 5;
    const int m0   = blockIdx.x * 16;
    const int n0   = (blockIdx.y * 4 + wv) * 16;

    const int mrow  = m0 + (lane & 15);        // A: M = lane&15 (both halves)
    const int khalf = (lane >> 4) * 8;         // A: K half select per lane group
    const int ncol  = n0 + (lane & 15);        // B: N = lane&15
    const int kb    = (lane >> 4) * 8;         // B: K half select per lane group

    v8f acc = {};
    #pragma unroll
    for (int k0 = 0; k0 < 64; k0 += 32) {
        // A fragment: two 8-float contiguous runs -> 4x b128 loads
        float4 p0 = *(const float4*)(x + mrow * 64 + k0 + khalf);
        float4 p1 = *(const float4*)(x + mrow * 64 + k0 + khalf + 4);
        float4 q0 = *(const float4*)(x + mrow * 64 + k0 + 16 + khalf);
        float4 q1 = *(const float4*)(x + mrow * 64 + k0 + 16 + khalf + 4);
        v16bf a;
        a[0]=(__bf16)p0.x; a[1]=(__bf16)p0.y; a[2]=(__bf16)p0.z; a[3]=(__bf16)p0.w;
        a[4]=(__bf16)p1.x; a[5]=(__bf16)p1.y; a[6]=(__bf16)p1.z; a[7]=(__bf16)p1.w;
        a[8]=(__bf16)q0.x; a[9]=(__bf16)q0.y; a[10]=(__bf16)q0.z; a[11]=(__bf16)q0.w;
        a[12]=(__bf16)q1.x; a[13]=(__bf16)q1.y; a[14]=(__bf16)q1.z; a[15]=(__bf16)q1.w;
        // B fragment: two 8-float contiguous runs -> 4x b128 loads
        const float* wp = w + ncol * 64 + k0 + kb;
        float4 b0 = *(const float4*)(wp);
        float4 b1 = *(const float4*)(wp + 4);
        float4 b2 = *(const float4*)(wp + 16);
        float4 b3 = *(const float4*)(wp + 20);
        v16bf b;
        b[0]=(__bf16)b0.x; b[1]=(__bf16)b0.y; b[2]=(__bf16)b0.z; b[3]=(__bf16)b0.w;
        b[4]=(__bf16)b1.x; b[5]=(__bf16)b1.y; b[6]=(__bf16)b1.z; b[7]=(__bf16)b1.w;
        b[8]=(__bf16)b2.x; b[9]=(__bf16)b2.y; b[10]=(__bf16)b2.z; b[11]=(__bf16)b2.w;
        b[12]=(__bf16)b3.x; b[13]=(__bf16)b3.y; b[14]=(__bf16)b3.z; b[15]=(__bf16)b3.w;
        acc = __builtin_amdgcn_wmma_f32_16x16x32_bf16(false, a, false, b,
                                                      (short)0, acc, false, false);
    }
    #pragma unroll
    for (int r = 0; r < 8; ++r) {
        int m = m0 + r + ((lane >> 4) * 8);
        int j = n0 + (lane & 15);
        float v = acc[r];
        if (j < DIN) xi[m * DIN + j]        = v;
        else         z [m * DIN + (j-DIN)]  = v;
    }
}

// ---------------------------------------------------------------------------
// K2: depthwise causal conv + SiLU -> xs; x_proj (N=36) via LDS dot;
//     dt_proj + softplus -> delta; split Bm / C.
// grid 65536 (one token), block 128 (one channel each)
// ---------------------------------------------------------------------------
__global__ void k_conv_proj(const float* __restrict__ xi,
                            const float* __restrict__ conv_w, const float* __restrict__ conv_b,
                            const float* __restrict__ xpw,
                            const float* __restrict__ dtw, const float* __restrict__ dtb,
                            float* __restrict__ xs, float* __restrict__ delta,
                            float* __restrict__ Bm, float* __restrict__ Cm) {
    const int bl = blockIdx.x;
    const int l  = bl & (LL - 1);
    const int d  = threadIdx.x;
    __shared__ float s_xs[DIN];
    __shared__ float s_db[36];

    float acc = conv_b[d];
    #pragma unroll
    for (int c = 0; c < 4; ++c) {
        int lc = l - 3 + c;
        float v = (lc >= 0) ? xi[(bl - 3 + c) * DIN + d] : 0.f;
        acc += conv_w[d * 4 + c] * v;
    }
    float sv = silu_f(acc);
    s_xs[d] = sv;
    xs[bl * DIN + d] = sv;
    __syncthreads();

    if (d < 36) {
        float s = 0.f;
        #pragma unroll 8
        for (int k = 0; k < DIN; ++k) s += s_xs[k] * xpw[d * DIN + k];
        s_db[d] = s;
        if (d >= 4 && d < 20)  Bm[bl * NST + (d - 4)]  = s;
        else if (d >= 20)      Cm[bl * NST + (d - 20)] = s;
    }
    __syncthreads();

    float dtv = dtb[d];
    #pragma unroll
    for (int r = 0; r < 4; ++r) dtv += s_db[r] * dtw[d * 4 + r];
    delta[bl * DIN + d] = softplus_f(dtv);
}

// ---------------------------------------------------------------------------
// K3a: chunk-local scan (h0 = 0). Stores raw y, end-state h, sum(delta).
// grid 1024 = (b, chunk); block 128. B/C chunk staged in LDS via TDM.
// ---------------------------------------------------------------------------
__global__ void k_scan_local(const float* __restrict__ delta, const float* __restrict__ xs,
                             const float* __restrict__ Bm, const float* __restrict__ Cm,
                             const float* __restrict__ A_log,
                             float* __restrict__ yraw, float* __restrict__ hloc,
                             float* __restrict__ Ssum) {
    const int b = blockIdx.x >> 6;
    const int c = blockIdx.x & 63;
    const int d = threadIdx.x;
    const int t0 = c * TC;

    float negA[NST];
    #pragma unroll
    for (int n = 0; n < NST; ++n) negA[n] = -__expf(A_log[d * NST + n]);

    __shared__ float sB[TC][NST];
    __shared__ float sC[TC][NST];
    const int base = b * LL + t0;
#if defined(HAVE_TDM)
    if (threadIdx.x < 32) {                       // one wave drives the TDM
        tdm_load_1d(&sB[0][0], &Bm[(size_t)base * NST], TC * NST);
        tdm_load_1d(&sC[0][0], &Cm[(size_t)base * NST], TC * NST);
        __builtin_amdgcn_s_wait_tensorcnt(0);
    }
#else
    for (int i = threadIdx.x; i < TC * NST; i += 128) {
        ((float*)sB)[i] = Bm[base * NST + i];
        ((float*)sC)[i] = Cm[base * NST + i];
    }
#endif
    __syncthreads();

    if (c + 1 < NCHUNK)   // warm L2 for the next chunk's shared B-projection
        __builtin_prefetch(&Bm[(base + TC) * NST], 0, 1);

    float h[NST];
    #pragma unroll
    for (int n = 0; n < NST; ++n) h[n] = 0.f;
    float S = 0.f;

    for (int t = 0; t < TC; ++t) {
        int idx = (base + t) * DIN + d;
        float dlt = delta[idx];
        float u   = xs[idx];
        S += dlt;
        float du = dlt * u;
        float yv = 0.f;
        #pragma unroll
        for (int n = 0; n < NST; ++n) {
            float dA = __expf(dlt * negA[n]);
            h[n] = dA * h[n] + du * sB[t][n];
            yv  += h[n] * sC[t][n];
        }
        yraw[idx] = yv;
    }
    int hbase = ((b * NCHUNK + c) * DIN + d) * NST;
    #pragma unroll
    for (int n = 0; n < NST; ++n) hloc[hbase + n] = h[n];
    Ssum[(b * NCHUNK + c) * DIN + d] = S;
}

// ---------------------------------------------------------------------------
// K3b: cross-chunk carry:  carry[c] = H_start(c);  H = exp(negA*S_c)*H + hloc_c
// one thread per (b, d, n) = 32768 threads, 64 sequential chunks each
// ---------------------------------------------------------------------------
__global__ void k_scan_carry(const float* __restrict__ hloc, const float* __restrict__ Ssum,
                             const float* __restrict__ A_log, float* __restrict__ carry) {
    int tid = blockIdx.x * blockDim.x + threadIdx.x;
    if (tid >= BB * DIN * NST) return;
    int n = tid & (NST - 1);
    int d = (tid >> 4) & (DIN - 1);
    int b = tid >> 11;
    float negA = -__expf(A_log[d * NST + n]);
    float H = 0.f;
    for (int c = 0; c < NCHUNK; ++c) {
        int hi = ((b * NCHUNK + c) * DIN + d) * NST + n;
        carry[hi] = H;
        H = __expf(negA * Ssum[(b * NCHUNK + c) * DIN + d]) * H + hloc[hi];
    }
}

// ---------------------------------------------------------------------------
// K3c: apply carry correction + D-skip + SiLU gate:
//   y_t = (yraw_t + sum_n C_t[n]*exp(negA_n*cumsumDelta_t)*carry[n] + u*D) * silu(z)
// grid 1024 = (b, chunk); block 128
// ---------------------------------------------------------------------------
__global__ void k_scan_fix(const float* __restrict__ delta, const float* __restrict__ xs,
                           const float* __restrict__ Cm, const float* __restrict__ A_log,
                           const float* __restrict__ Dp, const float* __restrict__ z,
                           const float* __restrict__ carry, float* __restrict__ y) {
    const int b = blockIdx.x >> 6;
    const int c = blockIdx.x & 63;
    const int d = threadIdx.x;
    const int base = b * LL + c * TC;

    float negA[NST], cv[NST];
    int hbase = ((b * NCHUNK + c) * DIN + d) * NST;
    #pragma unroll
    for (int n = 0; n < NST; ++n) {
        negA[n] = -__expf(A_log[d * NST + n]);
        cv[n]   = carry[hbase + n];
    }
    float Dd = Dp[d];

    __shared__ float sC[TC][NST];
#if defined(HAVE_TDM)
    if (threadIdx.x < 32) {
        tdm_load_1d(&sC[0][0], &Cm[(size_t)base * NST], TC * NST);
        __builtin_amdgcn_s_wait_tensorcnt(0);
    }
#else
    for (int i = threadIdx.x; i < TC * NST; i += 128)
        ((float*)sC)[i] = Cm[base * NST + i];
#endif
    __syncthreads();

    float cs = 0.f;
    for (int t = 0; t < TC; ++t) {
        int idx = (base + t) * DIN + d;
        float dlt = delta[idx];
        cs += dlt;
        float u = xs[idx];
        float corr = 0.f;
        #pragma unroll
        for (int n = 0; n < NST; ++n)
            corr += sC[t][n] * __expf(negA[n] * cs) * cv[n];
        float g = silu_f(z[idx]);
        y[idx] = (y[idx] + corr + u * Dd) * g;
    }
}

// ---------------------------------------------------------------------------
// K4: out_proj (bf16 WMMA, 16 WMMAs per 16-row tile) + LayerNorm + residual SiLU
// grid 1024, block 128 (4 waves); wave -> 16 rows x 64 cols
// ---------------------------------------------------------------------------
__global__ void k_out_ln(const float* __restrict__ y, const float* __restrict__ w,
                         const float* __restrict__ xin,
                         const float* __restrict__ ln_g, const float* __restrict__ ln_b,
                         float* __restrict__ out) {
    const int lane = threadIdx.x & 31;
    const int wv   = threadIdx.x >> 5;
    const int m0   = (blockIdx.x * 4 + wv) * 16;

    const int mrow  = m0 + (lane & 15);
    const int khalf = (lane >> 4) * 8;
    const int kb    = (lane >> 4) * 8;

    __shared__ float s_tile[4][16][65];   // +1 pad: conflict-free column reads

    v8f acc[4];
    #pragma unroll
    for (int nt = 0; nt < 4; ++nt) acc[nt] = (v8f){};

    #pragma unroll
    for (int k0 = 0; k0 < DIN; k0 += 32) {
        float4 p0 = *(const float4*)(y + mrow * DIN + k0 + khalf);
        float4 p1 = *(const float4*)(y + mrow * DIN + k0 + khalf + 4);
        float4 q0 = *(const float4*)(y + mrow * DIN + k0 + 16 + khalf);
        float4 q1 = *(const float4*)(y + mrow * DIN + k0 + 16 + khalf + 4);
        v16bf a;
        a[0]=(__bf16)p0.x; a[1]=(__bf16)p0.y; a[2]=(__bf16)p0.z; a[3]=(__bf16)p0.w;
        a[4]=(__bf16)p1.x; a[5]=(__bf16)p1.y; a[6]=(__bf16)p1.z; a[7]=(__bf16)p1.w;
        a[8]=(__bf16)q0.x; a[9]=(__bf16)q0.y; a[10]=(__bf16)q0.z; a[11]=(__bf16)q0.w;
        a[12]=(__bf16)q1.x; a[13]=(__bf16)q1.y; a[14]=(__bf16)q1.z; a[15]=(__bf16)q1.w;
        #pragma unroll
        for (int nt = 0; nt < 4; ++nt) {
            const float* wp = w + (nt * 16 + (lane & 15)) * DIN + k0 + kb;
            float4 b0 = *(const float4*)(wp);
            float4 b1 = *(const float4*)(wp + 4);
            float4 b2 = *(const float4*)(wp + 16);
            float4 b3 = *(const float4*)(wp + 20);
            v16bf b;
            b[0]=(__bf16)b0.x; b[1]=(__bf16)b0.y; b[2]=(__bf16)b0.z; b[3]=(__bf16)b0.w;
            b[4]=(__bf16)b1.x; b[5]=(__bf16)b1.y; b[6]=(__bf16)b1.z; b[7]=(__bf16)b1.w;
            b[8]=(__bf16)b2.x; b[9]=(__bf16)b2.y; b[10]=(__bf16)b2.z; b[11]=(__bf16)b2.w;
            b[12]=(__bf16)b3.x; b[13]=(__bf16)b3.y; b[14]=(__bf16)b3.z; b[15]=(__bf16)b3.w;
            acc[nt] = __builtin_amdgcn_wmma_f32_16x16x32_bf16(false, a, false, b,
                                                              (short)0, acc[nt], false, false);
        }
    }
    // scatter accumulators to LDS (ISA C/D layout: VGPR r -> row r / r+8 per lane half)
    #pragma unroll
    for (int nt = 0; nt < 4; ++nt)
        #pragma unroll
        for (int r = 0; r < 8; ++r)
            s_tile[wv][r + ((lane >> 4) * 8)][nt * 16 + (lane & 15)] = acc[nt][r];
    __syncthreads();

    if (lane < 16) {
        const int m = m0 + lane;
        float mu = 0.f;
        #pragma unroll 8
        for (int cidx = 0; cidx < DM; ++cidx) mu += s_tile[wv][lane][cidx];
        mu *= (1.f / DM);
        float var = 0.f;
        #pragma unroll 8
        for (int cidx = 0; cidx < DM; ++cidx) {
            float dv = s_tile[wv][lane][cidx] - mu;
            var += dv * dv;
        }
        var *= (1.f / DM);
        float rstd = rsqrtf(var + 1e-5f);
        #pragma unroll 8
        for (int cidx = 0; cidx < DM; ++cidx) {
            float ln = (s_tile[wv][lane][cidx] - mu) * rstd * ln_g[cidx] + ln_b[cidx];
            out[m * DM + cidx] = xin[m * DM + cidx] + silu_f(ln);
        }
    }
}

// ---------------------------------------------------------------------------
extern "C" void kernel_launch(void* const* d_in, const int* in_sizes, int n_in,
                              void* d_out, int out_size, void* d_ws, size_t ws_size,
                              hipStream_t stream) {
    const float* x         = (const float*)d_in[0];
    const float* in_proj_w = (const float*)d_in[1];
    const float* conv_w    = (const float*)d_in[2];
    const float* conv_b    = (const float*)d_in[3];
    const float* x_proj_w  = (const float*)d_in[4];
    const float* dt_proj_w = (const float*)d_in[5];
    const float* dt_proj_b = (const float*)d_in[6];
    const float* A_log     = (const float*)d_in[7];
    const float* Dp        = (const float*)d_in[8];
    const float* out_proj_w= (const float*)d_in[9];
    const float* ln_g      = (const float*)d_in[10];
    const float* ln_b      = (const float*)d_in[11];
    float* out = (float*)d_out;

    // workspace layout (floats)
    float* xi    = (float*)d_ws;                  // 65536*128
    float* z     = xi    + (size_t)NTOK * DIN;
    float* xs    = z     + (size_t)NTOK * DIN;
    float* delta = xs    + (size_t)NTOK * DIN;
    float* Bm    = delta + (size_t)NTOK * DIN;    // 65536*16
    float* Cm    = Bm    + (size_t)NTOK * NST;
    float* yraw  = Cm    + (size_t)NTOK * NST;    // 65536*128
    float* hloc  = yraw  + (size_t)NTOK * DIN;    // 16*64*128*16
    float* Ssum  = hloc  + (size_t)BB * NCHUNK * DIN * NST;  // 16*64*128
    float* carry = Ssum  + (size_t)BB * NCHUNK * DIN;        // 16*64*128*16

    k_in_proj  <<<dim3(NTOK / 16, 4), 128, 0, stream>>>(x, in_proj_w, xi, z);
    k_conv_proj<<<dim3(NTOK),        128, 0, stream>>>(xi, conv_w, conv_b, x_proj_w,
                                                       dt_proj_w, dt_proj_b,
                                                       xs, delta, Bm, Cm);
    k_scan_local<<<dim3(BB * NCHUNK), 128, 0, stream>>>(delta, xs, Bm, Cm, A_log,
                                                        yraw, hloc, Ssum);
    k_scan_carry<<<dim3((BB * DIN * NST + 255) / 256), 256, 0, stream>>>(hloc, Ssum,
                                                                         A_log, carry);
    k_scan_fix <<<dim3(BB * NCHUNK), 128, 0, stream>>>(delta, xs, Cm, A_log, Dp, z,
                                                       carry, yraw);
    k_out_ln   <<<dim3(NTOK / 64),   128, 0, stream>>>(yraw, out_proj_w, x,
                                                       ln_g, ln_b, out);
}
